// LSTMEDModule_65429531787766
// MI455X (gfx1250) — matrix operand order repaired and checked by reading, not compiled, not run on hardware
//
#include <hip/hip_runtime.h>

#define B_ 512
#define T_ 256
#define F_ 128
#define H_ 512
#define G_ (4 * H_)

typedef __bf16 bf16_t;
typedef bf16_t v16bf __attribute__((ext_vector_type(16)));
typedef float v8f __attribute__((ext_vector_type(8)));

__device__ __forceinline__ float sigmoidf_(float x) {
  return 1.0f / (1.0f + __expf(-x));
}

// fp32 -> bf16 round-to-nearest-even, as raw bits.
__device__ __forceinline__ unsigned short f2bf_rne(float f) {
  unsigned u = __float_as_uint(f);
  unsigned r = u + 0x7fffu + ((u >> 16) & 1u);
  return (unsigned short)(r >> 16);
}

// A fragment: 16x32 (MxK) bf16 from a bf16 matrix with row stride `ld` (elems).
// ISA layout (V_WMMA_*_16X16X32 A): lanes 0-15/16-31 both hold M=row; lane-half
// adds +8 within each K-group of 16; VGPR v: K = 16*(v>>2) + 2*(v&3) + 8*half.
// Per lane: two contiguous 8-element (16 B) runs -> two b128 loads.
__device__ __forceinline__ v16bf load_a_bf(const unsigned short* __restrict__ src, int ld,
                                           int row, int k0, int half) {
  const bf16_t* p = (const bf16_t*)(src + (size_t)row * ld) + k0 + half * 8;
  v16bf a;
#pragma unroll
  for (int e = 0; e < 8; ++e) a[e] = p[e];          // K = k0+8h .. +7
#pragma unroll
  for (int e = 0; e < 8; ++e) a[8 + e] = p[16 + e]; // K = k0+16+8h .. +7
  return a;
}

// B fragment: 32x16 (KxN) bf16 where B[k][n] = W[n][k]; W row-major bf16 with
// row stride `ld`.  Lane holds column n; same K order as A: elements 0-7 hold
// K = k0+8*half .. +7, elements 8-15 hold K = k0+16+8*half .. +7.
__device__ __forceinline__ v16bf load_b_bf(const unsigned short* __restrict__ W, int ld,
                                           int col, int k0, int half) {
  const bf16_t* p = (const bf16_t*)(W + (size_t)col * ld) + k0 + half * 8;
  v16bf b;
#pragma unroll
  for (int e = 0; e < 8; ++e) b[e] = p[e];
#pragma unroll
  for (int e = 0; e < 8; ++e) b[8 + e] = p[16 + e];
  return b;
}

// One fused LSTM step:  gates = x@W_ih^T + h@W_hh^T + bias ; cell update.
//
// Tiling for L2-traffic economy: each wave owns a 16(M) x 32(J) tile and
// accumulates all four gate blocks for both J-subtiles (4 x 2 = 8 fp32
// accumulator fragments).  The 8 waves of a block span 8 consecutive M-tiles
// with the SAME j-group, so every B (weight) fragment is identical across the
// block and hits in the WGP$; the weight matrix is pulled from L2 only once
// per M-block (4x) instead of once per M-tile (32x).
// Grid: (B/(16*8)) M-blocks x (H/32) J-groups = 4 x 16 = 64 blocks x 8 waves.
template <int KX>
__global__ __launch_bounds__(256)
void lstm_step_kernel(const unsigned short* __restrict__ x_bf, int x_ld,
                      const unsigned short* __restrict__ Wih_bf,
                      const unsigned short* __restrict__ Whh_bf,
                      const float* __restrict__ bias,
                      const unsigned short* __restrict__ h_in,
                      const float* __restrict__ c_in,
                      unsigned short* __restrict__ h_out,
                      float* __restrict__ c_out) {
  int wib  = threadIdx.x >> 5;     // wave in block: M-tile within M-block
  int lane = threadIdx.x & 31;
  int half = lane >> 4;
  int lrow = lane & 15;
  int mb = blockIdx.x >> 4;        // 4 M-blocks (8 M-tiles each)
  int jg = blockIdx.x & 15;        // 16 J-groups (32 hidden units each)
  int m0 = (mb * 8 + wib) << 4;    // 16 batch rows per wave
  int j0 = jg << 5;                // 32 hidden units per wave

  v8f acc[4][2] = {};              // [gate i,f,g,o][J-subtile]

  // input contribution: K = KX (=F), fully unrolled
#pragma unroll
  for (int k0 = 0; k0 < KX; k0 += 32) {
    v16bf a = load_a_bf(x_bf, x_ld, m0 + lrow, k0, half);
#pragma unroll
    for (int g = 0; g < 4; ++g) {
#pragma unroll
      for (int s = 0; s < 2; ++s) {
        v16bf b = load_b_bf(Wih_bf, KX, g * H_ + j0 + s * 16 + lrow, k0, half);
        acc[g][s] = __builtin_amdgcn_wmma_f32_16x16x32_bf16(
            false, a, false, b, (short)0, acc[g][s], false, false);
      }
    }
  }
  // recurrent contribution: K = H; prefetch next K-block of the static,
  // L2-resident weight matrix into the WGP cache (global_prefetch_b8)
  for (int k0 = 0; k0 < H_; k0 += 32) {
    if (k0 + 32 < H_) {
      __builtin_prefetch(Whh_bf + (size_t)(j0 + lrow) * H_ + k0 + 32, 0, 1);
      __builtin_prefetch(Whh_bf + (size_t)(2 * H_ + j0 + lrow) * H_ + k0 + 32, 0, 1);
    }
    v16bf a = load_a_bf(h_in, H_, m0 + lrow, k0, half);
#pragma unroll
    for (int g = 0; g < 4; ++g) {
#pragma unroll
      for (int s = 0; s < 2; ++s) {
        v16bf b = load_b_bf(Whh_bf, H_, g * H_ + j0 + s * 16 + lrow, k0, half);
        acc[g][s] = __builtin_amdgcn_wmma_f32_16x16x32_bf16(
            false, a, false, b, (short)0, acc[g][s], false, false);
      }
    }
  }

  // epilogue: fused LSTM cell update for both J-subtiles.
  // C/D layout: row = m0 + 8*half + v, col = j0 + 16*s + lane%16.
#pragma unroll
  for (int s = 0; s < 2; ++s) {
    int col = j0 + s * 16 + lrow;
    float bi = bias[0 * H_ + col];
    float bf = bias[1 * H_ + col];
    float bg = bias[2 * H_ + col];
    float bo = bias[3 * H_ + col];
#pragma unroll
    for (int v = 0; v < 8; ++v) {
      int row = m0 + half * 8 + v;
      size_t idx = (size_t)row * H_ + col;
      float ig = sigmoidf_(acc[0][s][v] + bi);
      float fg = sigmoidf_(acc[1][s][v] + bf);
      float gg = tanhf(acc[2][s][v] + bg);
      float og = sigmoidf_(acc[3][s][v] + bo);
      float c = fg * c_in[idx] + ig * gg;
      float h = og * tanhf(c);
      c_out[idx] = c;
      h_out[idx] = f2bf_rne(h);
    }
  }
}

// Decoder output projection: y = h @ Wo^T + bo.  Emits fp32 y to d_out at the
// flipped time index (T-1-t) and bf16 y for the next decoder GEMM.
// Grid: (B/16)*(F/16) = 256 waves = 32 blocks (block shares the A tile).
__global__ __launch_bounds__(256)
void out_proj_kernel(const unsigned short* __restrict__ h_bf,
                     const unsigned short* __restrict__ Wo_bf,
                     const float* __restrict__ bo,
                     unsigned short* __restrict__ y_bf,
                     float* __restrict__ out, int t) {
  int wave = blockIdx.x * (blockDim.x >> 5) + (threadIdx.x >> 5);
  int lane = threadIdx.x & 31;
  int half = lane >> 4;
  int lrow = lane & 15;
  int tileM = wave >> 3;           // B/16 = 32
  int tileN = wave & 7;            // F/16 = 8
  int m0 = tileM << 4;
  int n0 = tileN << 4;

  v8f acc = {};
#pragma unroll 4
  for (int k0 = 0; k0 < H_; k0 += 32) {
    v16bf a = load_a_bf(h_bf, H_, m0 + lrow, k0, half);
    v16bf b = load_b_bf(Wo_bf, H_, n0 + lrow, k0, half);
    acc = __builtin_amdgcn_wmma_f32_16x16x32_bf16(
        false, a, false, b, (short)0, acc, false, false);
  }

  int col = n0 + lrow;
  float bb = bo[col];
#pragma unroll
  for (int v = 0; v < 8; ++v) {
    int row = m0 + half * 8 + v;
    float yv = acc[v] + bb;
    out[(size_t)row * (T_ * F_) + (size_t)(T_ - 1 - t) * F_ + col] = yv;
    y_bf[(size_t)row * F_ + col] = f2bf_rne(yv);
  }
}

// fp32 -> bf16 (RNE) bulk conversion (weights once, ts_batch once).
__global__ void cvt_bf16_kernel(const float* __restrict__ src,
                                unsigned short* __restrict__ dst, int n) {
  int i = blockIdx.x * blockDim.x + threadIdx.x;
  if (i >= n) return;
  dst[i] = f2bf_rne(src[i]);
}

__global__ void add_vec_kernel(const float* __restrict__ a,
                               const float* __restrict__ b,
                               float* __restrict__ dst, int n) {
  int i = blockIdx.x * blockDim.x + threadIdx.x;
  if (i < n) dst[i] = a[i] + b[i];
}

__global__ void zero_f32_kernel(float* __restrict__ p, int n) {
  int i = blockIdx.x * blockDim.x + threadIdx.x;
  if (i < n) p[i] = 0.0f;
}

__global__ void zero_u16_kernel(unsigned short* __restrict__ p, int n) {
  int i = blockIdx.x * blockDim.x + threadIdx.x;
  if (i < n) p[i] = 0;   // bf16 +0.0
}

extern "C" void kernel_launch(void* const* d_in, const int* in_sizes, int n_in,
                              void* d_out, int out_size, void* d_ws, size_t ws_size,
                              hipStream_t stream) {
  const float* ts   = (const float*)d_in[0];   // [B,T,F]
  const float* eWih = (const float*)d_in[1];   // [4H,F]
  const float* eWhh = (const float*)d_in[2];   // [4H,H]
  const float* eBih = (const float*)d_in[3];   // [4H]
  const float* eBhh = (const float*)d_in[4];   // [4H]
  const float* dWih = (const float*)d_in[5];
  const float* dWhh = (const float*)d_in[6];
  const float* dBih = (const float*)d_in[7];
  const float* dBhh = (const float*)d_in[8];
  const float* Wo   = (const float*)d_in[9];   // [F,H]
  const float* bo   = (const float*)d_in[10];  // [F]
  float* out = (float*)d_out;

  // workspace layout (~41 MB total)
  char* w = (char*)d_ws;
  unsigned short* eWih_bf = (unsigned short*)w; w += (size_t)G_ * F_ * 2;
  unsigned short* eWhh_bf = (unsigned short*)w; w += (size_t)G_ * H_ * 2;
  unsigned short* dWih_bf = (unsigned short*)w; w += (size_t)G_ * F_ * 2;
  unsigned short* dWhh_bf = (unsigned short*)w; w += (size_t)G_ * H_ * 2;
  unsigned short* Wo_bf   = (unsigned short*)w; w += (size_t)F_ * H_ * 2;
  unsigned short* ts_bf   = (unsigned short*)w; w += (size_t)B_ * T_ * F_ * 2;
  float* eBias = (float*)w; w += (size_t)G_ * 4;
  float* dBias = (float*)w; w += (size_t)G_ * 4;
  unsigned short* hbuf[2];
  hbuf[0] = (unsigned short*)w; w += (size_t)B_ * H_ * 2;
  hbuf[1] = (unsigned short*)w; w += (size_t)B_ * H_ * 2;
  float* cbuf[2];
  cbuf[0] = (float*)w; w += (size_t)B_ * H_ * 4;
  cbuf[1] = (float*)w; w += (size_t)B_ * H_ * 4;
  unsigned short* ybuf = (unsigned short*)w; w += (size_t)B_ * F_ * 2;

  // one-time prep: bf16 weights + inputs, fused biases, zero initial state
  cvt_bf16_kernel<<<(G_ * F_ + 255) / 256, 256, 0, stream>>>(eWih, eWih_bf, G_ * F_);
  cvt_bf16_kernel<<<(G_ * H_ + 255) / 256, 256, 0, stream>>>(eWhh, eWhh_bf, G_ * H_);
  cvt_bf16_kernel<<<(G_ * F_ + 255) / 256, 256, 0, stream>>>(dWih, dWih_bf, G_ * F_);
  cvt_bf16_kernel<<<(G_ * H_ + 255) / 256, 256, 0, stream>>>(dWhh, dWhh_bf, G_ * H_);
  cvt_bf16_kernel<<<(F_ * H_ + 255) / 256, 256, 0, stream>>>(Wo, Wo_bf, F_ * H_);
  cvt_bf16_kernel<<<(B_ * T_ * F_ + 255) / 256, 256, 0, stream>>>(ts, ts_bf, B_ * T_ * F_);
  add_vec_kernel<<<(G_ + 255) / 256, 256, 0, stream>>>(eBih, eBhh, eBias, G_);
  add_vec_kernel<<<(G_ + 255) / 256, 256, 0, stream>>>(dBih, dBhh, dBias, G_);
  zero_u16_kernel<<<(B_ * H_ + 255) / 256, 256, 0, stream>>>(hbuf[0], B_ * H_);
  zero_f32_kernel<<<(B_ * H_ + 255) / 256, 256, 0, stream>>>(cbuf[0], B_ * H_);

  const int stepBlocks = (B_ / 128) * (H_ / 32);     // 4 M-blocks x 16 J-groups = 64
  const int projBlocks = (B_ / 16) * (F_ / 16) / 8;  // 32 blocks x 8 waves

  // encoder: 256 sequential fused GEMM+cell steps, ping-pong h/c
  for (int t = 0; t < T_; ++t) {
    int p = t & 1;
    lstm_step_kernel<F_><<<stepBlocks, 256, 0, stream>>>(
        ts_bf + (size_t)t * F_, T_ * F_,        // x_t view of bf16 [B,T,F]
        eWih_bf, eWhh_bf, eBias,
        hbuf[p], cbuf[p], hbuf[p ^ 1], cbuf[p ^ 1]);
  }
  // after an even number of steps the state is back in buffer 0;
  // decoder restarts with c = 0 (reference semantics)
  zero_f32_kernel<<<(B_ * H_ + 255) / 256, 256, 0, stream>>>(cbuf[0], B_ * H_);

  // decoder: emit y = Linear(h) (scattered to flipped time index), then step
  for (int t = 0; t < T_; ++t) {
    int p = t & 1;
    out_proj_kernel<<<projBlocks, 256, 0, stream>>>(hbuf[p], Wo_bf, bo, ybuf, out, t);
    lstm_step_kernel<F_><<<stepBlocks, 256, 0, stream>>>(
        ybuf, F_,
        dWih_bf, dWhh_bf, dBias,
        hbuf[p], cbuf[p], hbuf[p ^ 1], cbuf[p ^ 1]);
  }
}
